// AMMSEBasicBlock_62904091018007
// MI455X (gfx1250) — matrix operand order repaired and checked by reading, not compiled, not run on hardware
//
#include <hip/hip_runtime.h>
#include <hip/hip_bf16.h>
#include <math.h>

// ---------------------------------------------------------------------------
// AMM-SE basic block for MI455X (gfx1250), wave32 + WMMA.
//
// Dominant work: two GEMMs [12544 x 4096] x [4096 x 256] done with
// v_wmma_f32_16x16x32_f16. Softly-assigned codewords (softmax over 16) are
// generated on the fly into LDS; LUT tiles are DMA'd with
// global_load_async_to_lds_b128. Software-pipelined: WMMA(chunk i) co-executes
// with softmax(chunk i+1) and the async copy of LUT(chunk i+1), one
// workgroup barrier per chunk.
// ---------------------------------------------------------------------------

typedef __attribute__((ext_vector_type(16))) _Float16 v16h;
typedef __attribute__((ext_vector_type(8)))  _Float16 v8h;
typedef __attribute__((ext_vector_type(8)))  float    v8f;

// exact pointee type the async-LDS builtin expects (from hipcc diagnostic)
typedef int v4i_vs __attribute__((vector_size(16)));
typedef __attribute__((address_space(1))) v4i_vs* as1_v4i_ptr;
typedef __attribute__((address_space(3))) v4i_vs* as3_v4i_ptr;

#define BATCH 16
#define CIN   256
#define HW    784     // 28*28
#define KCB   16      // centroids per codebook
#define BN_TOTAL (BATCH * HW)   // 12544 GEMM rows (spatial)
#define MTILE 64                // spatial rows per workgroup

#if defined(__gfx1250__) && __has_builtin(__builtin_amdgcn_global_load_async_to_lds_b128)
#define USE_ASYNC_LDS 1
#else
#define USE_ASYNC_LDS 0
#endif

__device__ __forceinline__ void copy16_to_lds(_Float16* dst, const _Float16* src) {
#if USE_ASYNC_LDS
  __builtin_amdgcn_global_load_async_to_lds_b128(
      (as1_v4i_ptr)src, (as3_v4i_ptr)dst, 0, 0);
#else
  *(v8h*)dst = *(const v8h*)src;   // global_load_b128 + ds_store_b128
#endif
}

__device__ __forceinline__ void wait_lds_copies() {
#if USE_ASYNC_LDS
#if __has_builtin(__builtin_amdgcn_s_wait_asynccnt)
  __builtin_amdgcn_s_wait_asynccnt(0);
#else
  asm volatile("s_wait_asynccnt 0" ::: "memory");
#endif
#endif
}

// ---------------------------------------------------------------------------
// LUT precompute: lut[c][j][o] = sum_s centroids[c][j][s] * wsub[c][s][o]
// stored fp16 as lutT[cb][o][kk], cb = c>>2, kk = 16*(c&3) + j  (chunk of 4
// codebooks => contiguous 256x64 fp16 = 32KB block per GEMM K-chunk).
// ---------------------------------------------------------------------------
__global__ __launch_bounds__(256) void lut_kernel(
    const float* __restrict__ cent, const float* __restrict__ wsub,
    _Float16* __restrict__ lutT) {
  int idx = blockIdx.x * 256 + threadIdx.x;      // 64*256*64 = 1048576 total
  int kk = idx & 63;
  int o  = (idx >> 6) & 255;
  int cb = idx >> 14;
  int c  = cb * 4 + (kk >> 4);
  int j  = kk & 15;
  float a = 0.f;
#pragma unroll
  for (int s = 0; s < 9; ++s)
    a += cent[(c * KCB + j) * 9 + s] * wsub[(c * 9 + s) * CIN + o];
  lutT[idx] = (_Float16)a;
}

// ---------------------------------------------------------------------------
// Fused AMM conv3x3 + BN (+ReLU): per workgroup, 64 spatial rows x 256
// output channels. GEMM orientation: M = output channel, N = spatial row
// (so D-tile lanes stride along contiguous spatial memory).
// ---------------------------------------------------------------------------
__global__ __launch_bounds__(256, 1) void amm_block_kernel(
    const float* __restrict__ in, const float* __restrict__ cent,
    const _Float16* __restrict__ lutT,
    const float* __restrict__ gamma, const float* __restrict__ beta,
    const float* __restrict__ mean, const float* __restrict__ var,
    float* __restrict__ outp, int doRelu) {
  __shared__ __attribute__((aligned(16))) _Float16 lutS[2][256 * 64];  // 2x32KB
  __shared__ __attribute__((aligned(16))) _Float16 attnS[2][64 * 64];  // 2x8KB
  __shared__ float invS[256];
  __shared__ float biasS[256];

  const int tid  = threadIdx.x;
  const int lane = tid & 31;
  const int wv   = tid >> 5;          // 8 waves; wave owns channels [32w,32w+32)
  const bool laneLo = lane < 16;

  {  // BN constants
    float iv = gamma[tid] * rsqrtf(var[tid] + 1e-5f);
    invS[tid]  = iv;
    biasS[tid] = beta[tid] - mean[tid] * iv;
  }

  const int mBase = blockIdx.x * MTILE;
  const int mi = tid & 63;            // this thread's softmax row
  const int cl = tid >> 6;            // this thread's codebook-in-chunk (0..3)

  const int m  = mBase + mi;
  const int b  = m / HW;
  const int n  = m - b * HW;
  const int oh = n / 28;
  const int ow = n - oh * 28;

  int   poff[9];
  float pmask[9];
#pragma unroll
  for (int s = 0; s < 9; ++s) {
    int kh = s / 3, kw = s - kh * 3;
    int ih = oh + kh - 1, iw = ow + kw - 1;
    bool ok = ((unsigned)ih < 28u) && ((unsigned)iw < 28u);
    poff[s]  = ok ? (ih * 28 + iw) : 0;
    pmask[s] = ok ? 1.f : 0.f;
  }
  const float* inB = in + b * (CIN * HW);

  // issue async LUT tile copy for chunk `cb` into buffer `buf`
  auto stage_lut = [&](int cb, int buf) {
    const _Float16* src = lutT + cb * 16384;
#pragma unroll
    for (int i = 0; i < 8; ++i) {
      int off = i * 2048 + tid * 8;
      copy16_to_lds(&lutS[buf][off], &src[off]);
    }
  };

  // soft codeword assignment for (row mi, codebook cb*4+cl) -> 16 fp16
  auto stage_attn = [&](int cb, int buf) {
    const int c = cb * 4 + cl;
    const float* cp = cent + c * (KCB * 9);
    const float* ip = inB + c * HW;
    float p[9];
#pragma unroll
    for (int s = 0; s < 9; ++s) p[s] = pmask[s] * ip[poff[s]];
    float dist[16];
    float dmin = 3.4e38f;
#pragma unroll
    for (int j = 0; j < 16; ++j) {
      float a2 = 0.f;
#pragma unroll
      for (int s = 0; s < 9; ++s) {
        float t = p[s] - cp[j * 9 + s];
        a2 += t * t;
      }
      dist[j] = a2;
      dmin = fminf(dmin, a2);
    }
    float es = 0.f;
#pragma unroll
    for (int j = 0; j < 16; ++j) {
      float e = __expf(dmin - dist[j]);  // softmax(-d), max-shifted
      dist[j] = e;
      es += e;
    }
    float inv = 1.f / es;
    _Float16* arow = &attnS[buf][mi * 64 + cl * 16];
#pragma unroll
    for (int j = 0; j < 16; ++j) arow[j] = (_Float16)(dist[j] * inv);
  };

  v8f acc[8] = {};  // [ms][ns] -> 2 channel-subtiles x 4 spatial-subtiles

  // ---- prologue: prepare chunk 0 into buffer 0
  stage_lut(0, 0);
  stage_attn(0, 0);
  wait_lds_copies();
  __syncthreads();

  // ---- pipelined main loop: one barrier per K-chunk of 64
  for (int cb = 0; cb < 64; ++cb) {
    const int buf = cb & 1;

    if (cb < 63) stage_lut(cb + 1, buf ^ 1);  // async DMA overlaps WMMA+softmax

    const _Float16* lsrc = &lutS[buf][0];
    const _Float16* asrc = &attnS[buf][0];

    // 2 WMMA K-steps of 32 over this chunk (load all fragments, then 8 WMMAs)
#pragma unroll
    for (int t = 0; t < 2; ++t) {
      v16h afr[2];
#pragma unroll
      for (int ms = 0; ms < 2; ++ms) {  // A: 16x32 fp16, M = channel rows
        int orow = wv * 32 + ms * 16 + (lane & 15);
        int k0 = t * 32 + (laneLo ? 0 : 16);  // lanes0-15: K0..15; hi: K16..31
        v8h a0 = *(const v8h*)&lsrc[orow * 64 + k0];
        v8h a1 = *(const v8h*)&lsrc[orow * 64 + k0 + 8];
        afr[ms] = __builtin_shufflevector(a0, a1, 0, 1, 2, 3, 4, 5, 6, 7, 8, 9,
                                          10, 11, 12, 13, 14, 15);
      }
      v16h bfr[4];
#pragma unroll
      for (int ns = 0; ns < 4; ++ns) {  // B: 32x16 fp16, N = spatial cols
        int mrow = ns * 16 + (lane & 15);
        int kb = t * 32 + (laneLo ? 0 : 16);  // lanes0-15: K0..15; hi: K16..31
        v8h b0 = *(const v8h*)&asrc[mrow * 64 + kb];
        v8h b1 = *(const v8h*)&asrc[mrow * 64 + kb + 8];
        bfr[ns] = __builtin_shufflevector(b0, b1, 0, 1, 2, 3, 4, 5, 6, 7, 8, 9,
                                          10, 11, 12, 13, 14, 15);
      }
#pragma unroll
      for (int ns = 0; ns < 4; ++ns)
#pragma unroll
        for (int ms = 0; ms < 2; ++ms)
          acc[ms * 4 + ns] = __builtin_amdgcn_wmma_f32_16x16x32_f16(
              false, afr[ms], false, bfr[ns], (short)0, acc[ms * 4 + ns],
              false, false);
    }

    // softmax for next chunk co-executes with in-flight WMMAs (disjoint LDS)
    if (cb < 63) stage_attn(cb + 1, buf ^ 1);

    wait_lds_copies();
    __syncthreads();
  }

  // epilogue: BN (+ReLU), store to out[b][o][n] (lanes stride contiguous n)
#pragma unroll
  for (int ns = 0; ns < 4; ++ns) {
    int mcol = mBase + ns * 16 + (lane & 15);
    int bb = mcol / HW;
    int nn = mcol - bb * HW;
    float* op = outp + bb * (CIN * HW) + nn;
#pragma unroll
    for (int ms = 0; ms < 2; ++ms) {
      int ob = wv * 32 + ms * 16 + (laneLo ? 0 : 8);  // D: vgpr r -> M=r(+8 hi)
      v8f a = acc[ms * 4 + ns];
#pragma unroll
      for (int r = 0; r < 8; ++r) {
        int o = ob + r;
        float v = a[r] * invS[o] + biasS[o];
        if (doRelu) v = fmaxf(v, 0.f);
        op[o * HW] = v;
      }
    }
  }
}

// ---------------------------------------------------------------------------
// SE: global average pool, 2-layer MLP, scale + residual + relu
// ---------------------------------------------------------------------------
__global__ __launch_bounds__(256) void pool_kernel(const float* __restrict__ y,
                                                   float* __restrict__ s) {
  __shared__ float red[256];
  int tid = threadIdx.x;
  const float* p = y + (size_t)blockIdx.x * HW;   // blockIdx = b*256 + c
  float a = 0.f;
  for (int i = tid; i < HW; i += 256) a += p[i];
  red[tid] = a;
  __syncthreads();
  for (int off = 128; off > 0; off >>= 1) {
    if (tid < off) red[tid] += red[tid + off];
    __syncthreads();
  }
  if (tid == 0) s[blockIdx.x] = red[0] * (1.f / (float)HW);
}

__global__ __launch_bounds__(256) void se_kernel(
    const float* __restrict__ s, const float* __restrict__ w1,
    const float* __restrict__ b1, const float* __restrict__ w2,
    const float* __restrict__ b2, float* __restrict__ scale) {
  __shared__ float sv[256];
  __shared__ float hid[16];
  int bidx = blockIdx.x, tid = threadIdx.x;
  sv[tid] = s[bidx * 256 + tid];
  __syncthreads();
  if (tid < 16) {
    float a = b1[tid];
    for (int i = 0; i < 256; ++i) a += sv[i] * w1[i * 16 + tid];
    hid[tid] = fmaxf(a, 0.f);
  }
  __syncthreads();
  float a = b2[tid];
#pragma unroll
  for (int i = 0; i < 16; ++i) a += hid[i] * w2[i * 256 + tid];
  scale[bidx * 256 + tid] = 1.f / (1.f + __expf(-a));
}

__global__ __launch_bounds__(256) void finalize_kernel(
    const float* __restrict__ y2, const float* __restrict__ scale,
    const float* __restrict__ x, float* __restrict__ out) {
  int idx = blockIdx.x * 256 + threadIdx.x;   // exact grid: 16*256*784
  int bc = idx / HW;
  float v = y2[idx] * scale[bc] + x[idx];
  out[idx] = fmaxf(v, 0.f);
}

// ---------------------------------------------------------------------------
extern "C" void kernel_launch(void* const* d_in, const int* in_sizes, int n_in,
                              void* d_out, int out_size, void* d_ws,
                              size_t ws_size, hipStream_t stream) {
  (void)in_sizes; (void)n_in; (void)out_size; (void)ws_size;
  const float* x     = (const float*)d_in[0];
  const float* cent1 = (const float*)d_in[1];
  const float* wsub1 = (const float*)d_in[2];
  const float* bn1g  = (const float*)d_in[3];
  const float* bn1b  = (const float*)d_in[4];
  const float* bn1m  = (const float*)d_in[5];
  const float* bn1v  = (const float*)d_in[6];
  const float* cent2 = (const float*)d_in[7];
  const float* wsub2 = (const float*)d_in[8];
  const float* bn2g  = (const float*)d_in[9];
  const float* bn2b  = (const float*)d_in[10];
  const float* bn2m  = (const float*)d_in[11];
  const float* bn2v  = (const float*)d_in[12];
  const float* sew1  = (const float*)d_in[13];
  const float* seb1  = (const float*)d_in[14];
  const float* sew2  = (const float*)d_in[15];
  const float* seb2  = (const float*)d_in[16];
  float* out = (float*)d_out;

  char* ws = (char*)d_ws;
  _Float16* lutT1 = (_Float16*)(ws);                              // 2 MiB
  _Float16* lutT2 = (_Float16*)(ws + ((size_t)2 << 20));          // 2 MiB
  float* y1   = (float*)(ws + ((size_t)4 << 20));                 // 12.25 MiB
  float* y2   = (float*)(ws + ((size_t)18 << 20));                // 12.25 MiB
  float* pool = (float*)(ws + ((size_t)32 << 20));                // 16 KiB
  float* scl  = pool + BATCH * CIN;                               // 16 KiB

  lut_kernel<<<4096, 256, 0, stream>>>(cent1, wsub1, lutT1);
  lut_kernel<<<4096, 256, 0, stream>>>(cent2, wsub2, lutT2);

  amm_block_kernel<<<BN_TOTAL / MTILE, 256, 0, stream>>>(
      x, cent1, lutT1, bn1g, bn1b, bn1m, bn1v, y1, 1);
  amm_block_kernel<<<BN_TOTAL / MTILE, 256, 0, stream>>>(
      y1, cent2, lutT2, bn2g, bn2b, bn2m, bn2v, y2, 0);

  pool_kernel<<<BATCH * CIN, 256, 0, stream>>>(y2, pool);
  se_kernel<<<BATCH, 256, 0, stream>>>(pool, sew1, seb1, sew2, seb2, scl);
  finalize_kernel<<<BN_TOTAL, 256, 0, stream>>>(y2, scl, x, out);
}
